// HDGODE_20933670601185
// MI455X (gfx1250) — compile-verified, matrix operand order checked by reading, not run on hardware
//
#include <hip/hip_runtime.h>

// ---------------- problem constants ----------------
#define T_STEPS 64
#define NB 16
#define NP 16
#define NJ 22

typedef __attribute__((ext_vector_type(16))) _Float16 v16h;
typedef __attribute__((ext_vector_type(8)))  _Float16 v8h;
typedef __attribute__((ext_vector_type(8)))  float    v8f;

// f16 weight arena offsets (in halfs); each matrix stored fragment-swizzled:
// idx = ((nt*chunks + ck)*32 + lane)*16 + e   (nt: 16-col tile, ck: 32-K chunk)
#define OFF_ODE1 0        // 4 * (64*64)
#define OFF_ODE2 16384    // 128*64 + 3*64*64 = 20480
#define OFF_ODE3 36864    // 20480
#define OFF_GRU1 57344    // b0: 3*(96*64)=18432 ; b1: 3*(128*64)=24576
#define OFF_GRU2 100352   // b0: 3*(160*64)=30720 ; b1: 24576
#define OFF_GRU3 155648
#define ARENA_HALFS 210944

struct DevParams {
  const float* xs; const float* ms; const float* h0;
  const float* s_adj1; const float* s_adj2; const float* s_adj3;
  const float* d_adj1; const float* d_adj2; const float* d_adj3;
  const float* odeW[3][4]; const float* odeB[3][4];
  const float* gruW[3][2][3];   // [level][block][r,u,c]
  const float* gruB[3][2][3];   // [level][block][r,u,c]
  const float* outW; const float* outB;
  _Float16* wa;                 // f16 weight arena (in d_ws)
  float* s1; float* s2; float* s3; // state (B,P,64) (B,P,5,64) (B,P,16,64)
  float* out;                   // (B,T,P,J,3)
};

// ---------------- fast activations (CDNA5 v_tanh_f32 TRANS op) ----------------
__device__ __forceinline__ float fast_tanh(float x) {
#if __has_builtin(__builtin_amdgcn_tanhf)
  return __builtin_amdgcn_tanhf(x);
#else
  return tanhf(x);
#endif
}
__device__ __forceinline__ float fast_sig(float x) {
  return 0.5f * fast_tanh(0.5f * x) + 0.5f;
}

// ---------------- WMMA fragment helpers (gfx1250, wave32) ----------------
// A (16xK f16, row-major LDS): lane L<16 -> row M=L, K {0..7}u{16..23}; L+16 -> +8.
// Per lane this is two contiguous 8-half runs -> two ds_load_b128.
__device__ __forceinline__ v16h load_fragA(const _Float16* base, int lda) {
  int l = threadIdx.x & 31; int m = l & 15; int hi = (l >> 4) & 1;
  const _Float16* r = base + m * lda + hi * 8;
  v8h lo  = *(const v8h*)(r);
  v8h hi8 = *(const v8h*)(r + 16);
  v16h a;
#pragma unroll
  for (int j = 0; j < 8; ++j) { a[j] = lo[j]; a[8 + j] = hi8[j]; }
  return a;
}
// B fragments are pre-swizzled in the arena: each lane's 16 values contiguous.
// C = A(16xKPAD) * B(KPAD x 64); wave w (of 4) computes cols [16w, 16w+16)
template <int KPAD>
__device__ __forceinline__ void wmma_gemm(const _Float16* A, const _Float16* Wsw,
                                          v8f& acc) {
  constexpr int CH = KPAD / 32;
  int w = (threadIdx.x >> 5) & 3;
  int lane = threadIdx.x & 31;
#pragma unroll
  for (int ck = 0; ck < CH; ++ck) {
    v16h a = load_fragA(A + ck * 32, KPAD);
    v16h b = *(const v16h*)(Wsw + ((w * CH + ck) << 9) + (lane << 4));
    acc = __builtin_amdgcn_wmma_f32_16x16x32_f16(false, a, false, b,
                                                 (short)0, acc, false, false);
  }
}
#define ACT_NONE 0
#define ACT_TANH 1
#define ACT_SIG  2
__device__ __forceinline__ void store_acc(float* H, const v8f& acc,
                                          const float* bias, int act) {
  int w = (threadIdx.x >> 5) & 3;
  int l = threadIdx.x & 31;
  int n = (l & 15) + w * 16;
  int hi = (l >> 4) & 1;
  float bv = bias ? bias[n] : 0.f;
#pragma unroll
  for (int r = 0; r < 8; ++r) {
    float v = acc[r] + bv;
    if (act == ACT_TANH) v = fast_tanh(v);
    else if (act == ACT_SIG) v = fast_sig(v);
    H[(r + hi * 8) * 64 + n] = v;
  }
}

// ---------------- small adjacency matmuls (VALU) ----------------
// T(16xC f16) = adj(RxR) @ X(RxC f32); rows >= R zero-filled.
__device__ __forceinline__ void adjmm_dense(const float* adj, int R, const float* X,
                                            int ldx, int C, _Float16* T, int ldt) {
  for (int idx = threadIdx.x; idx < 16 * C; idx += blockDim.x) {
    int i = idx / C, k = idx - i * C;
    float s = 0.f;
    if (i < R)
      for (int j = 0; j < R; ++j) s += adj[i * R + j] * X[j * ldx + k];
    T[i * ldt + k] = (_Float16)s;
  }
}
// block-diagonal 4 groups of 4x4 (R=16)
__device__ __forceinline__ void adjmm_bdiag(const float* adj, const float* X,
                                            int ldx, int C, _Float16* T, int ldt) {
  for (int idx = threadIdx.x; idx < 16 * C; idx += blockDim.x) {
    int i = idx / C, k = idx - i * C;
    int g = i >> 2, ii = i & 3;
    float s = 0.f;
#pragma unroll
    for (int q = 0; q < 4; ++q) s += adj[(g * 4 + ii) * 4 + q] * X[(g * 4 + q) * ldx + k];
    T[i * ldt + k] = (_Float16)s;
  }
}

// ---------------- fused GRU cell core ----------------
// X: f32 16xKPAD, cols [0,dx)=x, [dx,dx+64)=h, rest zero.  hS: f32 16x64.
// Hout = u*h + (1-u)*tanh(adj@[x, r*h] @ Wc + bc).  Hout may alias Cb.
template <int KPAD, int ADJMODE>
__device__ void gru_core(int R, int dx, const float* adj,
                         float* X, _Float16* T,
                         const _Float16* Wr, const _Float16* Wu, const _Float16* Wc,
                         const float* br, const float* bu, const float* bc,
                         const float* hS, float* Rb, float* Ub, float* Cb, float* Hout) {
  __syncthreads();
  if (ADJMODE) adjmm_bdiag(adj, X, KPAD, KPAD, T, KPAD);
  else         adjmm_dense(adj, R, X, KPAD, KPAD, T, KPAD);
  __syncthreads();
  v8f accr = {}; wmma_gemm<KPAD>(T, Wr, accr); store_acc(Rb, accr, br, ACT_SIG);
  v8f accu = {}; wmma_gemm<KPAD>(T, Wu, accu); store_acc(Ub, accu, bu, ACT_SIG);
  __syncthreads();
  for (int idx = threadIdx.x; idx < 16 * 64; idx += blockDim.x) {
    int i = idx >> 6, k = idx & 63;
    X[i * KPAD + dx + k] = Rb[idx] * hS[idx];
  }
  __syncthreads();
  if (ADJMODE) adjmm_bdiag(adj, X, KPAD, KPAD, T, KPAD);
  else         adjmm_dense(adj, R, X, KPAD, KPAD, T, KPAD);
  __syncthreads();
  v8f accc = {}; wmma_gemm<KPAD>(T, Wc, accc); store_acc(Cb, accc, bc, ACT_TANH);
  __syncthreads();
  for (int idx = threadIdx.x; idx < 16 * 64; idx += blockDim.x)
    Hout[idx] = Ub[idx] * hS[idx] + (1.f - Ub[idx]) * Cb[idx];
  __syncthreads();
}

// ---------------- weight convert+swizzle (f32 -> fragment-major f16) ----------------
__global__ __launch_bounds__(256) void k_wconv(DevParams P) {
  int id = blockIdx.x;
  const float* src = nullptr; _Float16* dst = nullptr; int rows = 0, pad = 0;
  const int Ro[4] = {128, 64, 64, 64};
  const int Oo[4] = {0, 8192, 12288, 16384};
  if (id < 4)        { src = P.odeW[0][id];     rows = 64;      pad = 64;
                       dst = P.wa + OFF_ODE1 + id * 4096; }
  else if (id < 8)   { int l = id - 4; src = P.odeW[1][l]; rows = Ro[l]; pad = Ro[l];
                       dst = P.wa + OFF_ODE2 + Oo[l]; }
  else if (id < 12)  { int l = id - 8; src = P.odeW[2][l]; rows = Ro[l]; pad = Ro[l];
                       dst = P.wa + OFF_ODE3 + Oo[l]; }
  else {
    int q = id - 12; int g = q / 6, r6 = q - g * 6; int blk = r6 / 3, wi = r6 - blk * 3;
    const int base[3] = {OFF_GRU1, OFF_GRU2, OFF_GRU3};
    int pad0 = (g == 0) ? 96 : 160;
    int rows0 = (g == 0) ? 66 : 130;
    if (blk == 0) { rows = rows0; pad = pad0; dst = P.wa + base[g] + wi * pad0 * 64; }
    else          { rows = 128;  pad = 128;  dst = P.wa + base[g] + 3 * pad0 * 64 + wi * 8192; }
    src = P.gruW[g][blk][wi];
  }
  int chunks = pad >> 5;
  for (int idx = threadIdx.x; idx < pad * 64; idx += blockDim.x) {
    int e = idx & 15;
    int lane = (idx >> 4) & 31;
    int t2 = idx >> 9;            // nt*chunks + ck
    int ck = t2 % chunks;
    int nt = t2 / chunks;
    int hi = lane >> 4, j = e >> 1, odd = e & 1;
    int k = ck * 32 + ((j < 4) ? 2 * j : 16 + 2 * (j - 4)) + hi * 8 + odd;
    int n = nt * 16 + (lane & 15);
    dst[idx] = (k < rows) ? (_Float16)src[k * 64 + n] : (_Float16)0.f;
  }
}

// ---------------- state init: broadcast h0 ----------------
__global__ __launch_bounds__(256) void k_init(DevParams P) {
  for (size_t i = blockIdx.x * blockDim.x + threadIdx.x; i < 360448;
       i += (size_t)gridDim.x * blockDim.x) {
    float v = P.h0[i & 63];
    if (i < 16384)                P.s1[i] = v;
    else if (i < 16384 + 81920)   P.s2[i - 16384] = v;
    else                          P.s3[i - 16384 - 81920] = v;
  }
}

// ---------------- ode1: per-b, 4 layers, s1 += dgnn(s1)*0.1 ----------------
__global__ __launch_bounds__(128) void k_ode1(DevParams P) {
  extern __shared__ __align__(32) char smem[];
  char* cp = smem;
  _Float16* W  = (_Float16*)cp;            cp += 16384 * 2;
  float*    adj = (float*)cp;              cp += 256 * 4;
  float*    Ha  = (float*)cp;              cp += 1024 * 4;
  float*    Hb  = (float*)cp;              cp += 1024 * 4;
  _Float16* T   = (_Float16*)cp;           cp += 1024 * 2;
  int b = blockIdx.x;
  for (int i = threadIdx.x; i < 16384; i += blockDim.x) W[i] = P.wa[OFF_ODE1 + i];
  for (int i = threadIdx.x; i < 256; i += blockDim.x)   adj[i] = P.s_adj1[i];
  float* s1 = P.s1 + (size_t)b * 1024;
  for (int i = threadIdx.x; i < 1024; i += blockDim.x)  Ha[i] = s1[i];
  __syncthreads();
  float* H = Ha; float* H2 = Hb;
#pragma unroll
  for (int l = 0; l < 4; ++l) {
    adjmm_dense(adj, 16, H, 64, 64, T, 64);
    __syncthreads();
    v8f acc = {}; wmma_gemm<64>(T, W + l * 4096, acc);
    store_acc(H2, acc, P.odeB[0][l], (l < 3) ? ACT_TANH : ACT_NONE);
    __syncthreads();
    float* tmp = H; H = H2; H2 = tmp;
  }
  for (int i = threadIdx.x; i < 1024; i += blockDim.x) s1[i] += H[i] * 0.1f;
}

// ---------------- ode2+ode3 fused per (b,p) ----------------
__global__ __launch_bounds__(128) void k_ode23(DevParams P) {
  extern __shared__ __align__(32) char smem[];
  char* cp = smem;
  _Float16* Wall = (_Float16*)cp;          cp += 40960 * 2;   // ode2 then ode3
  float* adj2 = (float*)cp;                cp += 32 * 4;
  float* adj3 = (float*)cp;                cp += 64 * 4;
  float* h11  = (float*)cp;                cp += 64 * 4;
  float* S2   = (float*)cp;                cp += 320 * 4;
  float* S3   = (float*)cp;                cp += 1024 * 4;
  float* X    = (float*)cp;                cp += 2048 * 4;    // 16x128
  _Float16* T = (_Float16*)cp;             cp += 2048 * 2;    // 16x128
  float* Ha   = (float*)cp;                cp += 1024 * 4;
  float* Hb   = (float*)cp;                cp += 1024 * 4;
  const _Float16* W2 = Wall;
  const _Float16* W3 = Wall + 20480;
  int bp = blockIdx.x;
  for (int i = threadIdx.x; i < 40960; i += blockDim.x) Wall[i] = P.wa[OFF_ODE2 + i];
  for (int i = threadIdx.x; i < 25; i += blockDim.x)    adj2[i] = P.s_adj2[i];
  for (int i = threadIdx.x; i < 64; i += blockDim.x)    adj3[i] = P.s_adj3[i];
  for (int i = threadIdx.x; i < 64; i += blockDim.x)    h11[i] = P.s1[(size_t)bp * 64 + i];
  for (int i = threadIdx.x; i < 320; i += blockDim.x)   S2[i] = P.s2[(size_t)bp * 320 + i];
  for (int i = threadIdx.x; i < 1024; i += blockDim.x)  S3[i] = P.s3[(size_t)bp * 1024 + i];
  __syncthreads();
  const int Ol[4] = {0, 8192, 12288, 16384};
  for (int it = 0; it < 2; ++it) {
    // ----- ode2 -----
    for (int idx = threadIdx.x; idx < 5 * 128; idx += blockDim.x) {
      int r = idx >> 7, c = idx & 127;
      X[r * 128 + c] = (c < 64) ? h11[c] : S2[r * 64 + (c - 64)];
    }
    __syncthreads();
    adjmm_dense(adj2, 5, X, 128, 128, T, 128);
    __syncthreads();
    float* H = Ha; float* H2 = Hb;
    { v8f a = {}; wmma_gemm<128>(T, W2, a); store_acc(H, a, P.odeB[1][0], ACT_TANH); }
    __syncthreads();
#pragma unroll
    for (int l = 1; l < 4; ++l) {
      adjmm_dense(adj2, 5, H, 64, 64, T, 64);
      __syncthreads();
      v8f a = {}; wmma_gemm<64>(T, W2 + Ol[l], a);
      store_acc(H2, a, P.odeB[1][l], (l < 3) ? ACT_TANH : ACT_NONE);
      __syncthreads();
      float* tmp = H; H = H2; H2 = tmp;
    }
    for (int idx = threadIdx.x; idx < 320; idx += blockDim.x) S2[idx] += H[idx] * 0.05f;
    __syncthreads();
    // ----- ode3 x2 -----
    for (int jt = 0; jt < 2; ++jt) {
      for (int idx = threadIdx.x; idx < 16 * 128; idx += blockDim.x) {
        int r = idx >> 7, c = idx & 127;
        X[idx] = (c < 64) ? S2[(1 + (r >> 2)) * 64 + c] : S3[r * 64 + (c - 64)];
      }
      __syncthreads();
      adjmm_bdiag(adj3, X, 128, 128, T, 128);
      __syncthreads();
      float* H = Ha; float* H2 = Hb;
      { v8f a = {}; wmma_gemm<128>(T, W3, a); store_acc(H, a, P.odeB[2][0], ACT_TANH); }
      __syncthreads();
#pragma unroll
      for (int l = 1; l < 4; ++l) {
        adjmm_bdiag(adj3, H, 64, 64, T, 64);
        __syncthreads();
        v8f a = {}; wmma_gemm<64>(T, W3 + Ol[l], a);
        store_acc(H2, a, P.odeB[2][l], (l < 3) ? ACT_TANH : ACT_NONE);
        __syncthreads();
        float* tmp = H; H = H2; H2 = tmp;
      }
      for (int idx = threadIdx.x; idx < 1024; idx += blockDim.x) S3[idx] += H[idx] * 0.025f;
      __syncthreads();
    }
  }
  for (int i = threadIdx.x; i < 320; i += blockDim.x)  P.s2[(size_t)bp * 320 + i] = S2[i];
  for (int i = threadIdx.x; i < 1024; i += blockDim.x) P.s3[(size_t)bp * 1024 + i] = S3[i];
}

// ---------------- gru1: per-b (rows = p) ----------------
__global__ __launch_bounds__(128) void k_gru1(DevParams P, int t) {
  extern __shared__ __align__(32) char smem[];
  char* cp = smem;
  _Float16* Wg = (_Float16*)cp;            cp += 43008 * 2;
  float* adj = (float*)cp;                 cp += 256 * 4;
  float* h1  = (float*)cp;                 cp += 1024 * 4;
  float* X   = (float*)cp;                 cp += 2048 * 4;   // 16x128 max
  _Float16* T = (_Float16*)cp;             cp += 2048 * 2;
  float* Rb  = (float*)cp;                 cp += 1024 * 4;
  float* Ub  = (float*)cp;                 cp += 1024 * 4;
  float* Cb  = (float*)cp;                 cp += 1024 * 4;
  float* Hx  = (float*)cp;                 cp += 1024 * 4;
  int b = blockIdx.x;
  for (int i = threadIdx.x; i < 43008; i += blockDim.x) Wg[i] = P.wa[OFF_GRU1 + i];
  for (int i = threadIdx.x; i < 256; i += blockDim.x)
    adj[i] = P.d_adj1[((size_t)b * T_STEPS + t) * 256 + i];
  for (int i = threadIdx.x; i < 1024; i += blockDim.x) h1[i] = P.s1[(size_t)b * 1024 + i];
  __syncthreads();
  // cell0: Kpad=96, x = masked xs joint0 (2 cols)
  for (int idx = threadIdx.x; idx < 16 * 96; idx += blockDim.x) {
    int i = idx / 96, c = idx - i * 96;
    float v = 0.f;
    if (c < 2) {
      size_t e = ((((size_t)b * T_STEPS + t) * NP + i) * NJ + 0);
      v = P.xs[e * 2 + c] * P.ms[e];
    } else if (c < 66) v = h1[i * 64 + (c - 2)];
    X[idx] = v;
  }
  gru_core<96, 0>(16, 2, adj, X, T, Wg, Wg + 6144, Wg + 12288,
                  P.gruB[0][0][0], P.gruB[0][0][1], P.gruB[0][0][2],
                  h1, Rb, Ub, Cb, Hx);
  // cell1: Kpad=128, x = Hx
  for (int idx = threadIdx.x; idx < 16 * 128; idx += blockDim.x) {
    int i = idx >> 7, c = idx & 127;
    X[idx] = (c < 64) ? Hx[i * 64 + c] : h1[i * 64 + (c - 64)];
  }
  gru_core<128, 0>(16, 64, adj, X, T, Wg + 18432, Wg + 26624, Wg + 34816,
                   P.gruB[0][1][0], P.gruB[0][1][1], P.gruB[0][1][2],
                   h1, Rb, Ub, Cb, Cb);
  for (int i = threadIdx.x; i < 1024; i += blockDim.x) P.s1[(size_t)b * 1024 + i] = Cb[i];
  // output projection, joint 0, rows = p
  for (int idx = threadIdx.x; idx < 16 * 3; idx += blockDim.x) {
    int r = idx / 3, o = idx - r * 3;
    float s = P.outB[o];
    for (int k = 0; k < 64; ++k) s += Cb[r * 64 + k] * P.outW[k * 3 + o];
    P.out[((((size_t)b * T_STEPS + t) * NP + r) * NJ + 0) * 3 + o] = s;
  }
}

// ---------------- gru2: per (b,p), rows = 5 ----------------
__global__ __launch_bounds__(128) void k_gru2(DevParams P, int t) {
  extern __shared__ __align__(32) char smem[];
  char* cp = smem;
  _Float16* Wg = (_Float16*)cp;            cp += 55296 * 2;
  float* adj = (float*)cp;                 cp += 32 * 4;
  float* hS  = (float*)cp;                 cp += 1024 * 4;
  float* h21 = (float*)cp;                 cp += 64 * 4;
  float* X   = (float*)cp;                 cp += 2560 * 4;   // 16x160
  _Float16* T = (_Float16*)cp;             cp += 2560 * 2;
  float* Rb  = (float*)cp;                 cp += 1024 * 4;
  float* Ub  = (float*)cp;                 cp += 1024 * 4;
  float* Cb  = (float*)cp;                 cp += 1024 * 4;
  float* Hx  = (float*)cp;                 cp += 1024 * 4;
  int bp = blockIdx.x; int b = bp >> 4, p = bp & 15;
  for (int i = threadIdx.x; i < 55296; i += blockDim.x) Wg[i] = P.wa[OFF_GRU2 + i];
  for (int i = threadIdx.x; i < 25; i += blockDim.x)
    adj[i] = P.d_adj2[((size_t)bp * T_STEPS + t) * 25 + i];
  for (int i = threadIdx.x; i < 1024; i += blockDim.x)
    hS[i] = (i < 320) ? P.s2[(size_t)bp * 320 + i] : 0.f;
  for (int i = threadIdx.x; i < 64; i += blockDim.x) h21[i] = P.s1[(size_t)bp * 64 + i];
  __syncthreads();
  // cell0: Kpad=160, dx=66, x = [h2_1 | x2]
  for (int idx = threadIdx.x; idx < 16 * 160; idx += blockDim.x) {
    int i = idx / 160, c = idx - i * 160;
    float v = 0.f;
    if (i < 5) {
      if (c < 64) v = h21[c];
      else if (c < 66) {
        size_t e = ((((size_t)b * T_STEPS + t) * NP + p) * NJ + (1 + i));
        v = P.xs[e * 2 + (c - 64)] * P.ms[e];
      } else if (c < 130) v = hS[i * 64 + (c - 66)];
    }
    X[idx] = v;
  }
  gru_core<160, 0>(5, 66, adj, X, T, Wg, Wg + 10240, Wg + 20480,
                   P.gruB[1][0][0], P.gruB[1][0][1], P.gruB[1][0][2],
                   hS, Rb, Ub, Cb, Hx);
  // cell1: Kpad=128, x = Hx
  for (int idx = threadIdx.x; idx < 16 * 128; idx += blockDim.x) {
    int i = idx >> 7, c = idx & 127;
    float v = 0.f;
    if (i < 5) v = (c < 64) ? Hx[i * 64 + c] : hS[i * 64 + (c - 64)];
    X[idx] = v;
  }
  gru_core<128, 0>(5, 64, adj, X, T, Wg + 30720, Wg + 38912, Wg + 47104,
                   P.gruB[1][1][0], P.gruB[1][1][1], P.gruB[1][1][2],
                   hS, Rb, Ub, Cb, Cb);
  for (int i = threadIdx.x; i < 320; i += blockDim.x) P.s2[(size_t)bp * 320 + i] = Cb[i];
  for (int idx = threadIdx.x; idx < 5 * 3; idx += blockDim.x) {
    int r = idx / 3, o = idx - r * 3;
    float s = P.outB[o];
    for (int k = 0; k < 64; ++k) s += Cb[r * 64 + k] * P.outW[k * 3 + o];
    P.out[((((size_t)b * T_STEPS + t) * NP + p) * NJ + (1 + r)) * 3 + o] = s;
  }
}

// ---------------- gru3: per (b,p), rows = 16 (block-diag adj) ----------------
__global__ __launch_bounds__(128) void k_gru3(DevParams P, int t) {
  extern __shared__ __align__(32) char smem[];
  char* cp = smem;
  _Float16* Wg = (_Float16*)cp;            cp += 55296 * 2;
  float* adj = (float*)cp;                 cp += 64 * 4;
  float* hS  = (float*)cp;                 cp += 1024 * 4;
  float* h22 = (float*)cp;                 cp += 320 * 4;
  float* X   = (float*)cp;                 cp += 2560 * 4;
  _Float16* T = (_Float16*)cp;             cp += 2560 * 2;
  float* Rb  = (float*)cp;                 cp += 1024 * 4;
  float* Ub  = (float*)cp;                 cp += 1024 * 4;
  float* Cb  = (float*)cp;                 cp += 1024 * 4;
  float* Hx  = (float*)cp;                 cp += 1024 * 4;
  int bp = blockIdx.x; int b = bp >> 4, p = bp & 15;
  for (int i = threadIdx.x; i < 55296; i += blockDim.x) Wg[i] = P.wa[OFF_GRU3 + i];
  for (int i = threadIdx.x; i < 64; i += blockDim.x)
    adj[i] = P.d_adj3[((size_t)bp * T_STEPS + t) * 64 + i];
  for (int i = threadIdx.x; i < 1024; i += blockDim.x) hS[i] = P.s3[(size_t)bp * 1024 + i];
  for (int i = threadIdx.x; i < 320; i += blockDim.x) h22[i] = P.s2[(size_t)bp * 320 + i];
  __syncthreads();
  // cell0: Kpad=160, dx=66, x = [h2_2[1+g] | x3]
  for (int idx = threadIdx.x; idx < 16 * 160; idx += blockDim.x) {
    int i = idx / 160, c = idx - i * 160;
    float v = 0.f;
    if (c < 64) v = h22[(1 + (i >> 2)) * 64 + c];
    else if (c < 66) {
      size_t e = ((((size_t)b * T_STEPS + t) * NP + p) * NJ + (6 + i));
      v = P.xs[e * 2 + (c - 64)] * P.ms[e];
    } else if (c < 130) v = hS[i * 64 + (c - 66)];
    X[idx] = v;
  }
  gru_core<160, 1>(16, 66, adj, X, T, Wg, Wg + 10240, Wg + 20480,
                   P.gruB[2][0][0], P.gruB[2][0][1], P.gruB[2][0][2],
                   hS, Rb, Ub, Cb, Hx);
  // cell1: Kpad=128
  for (int idx = threadIdx.x; idx < 16 * 128; idx += blockDim.x) {
    int i = idx >> 7, c = idx & 127;
    X[idx] = (c < 64) ? Hx[i * 64 + c] : hS[i * 64 + (c - 64)];
  }
  gru_core<128, 1>(16, 64, adj, X, T, Wg + 30720, Wg + 38912, Wg + 47104,
                   P.gruB[2][1][0], P.gruB[2][1][1], P.gruB[2][1][2],
                   hS, Rb, Ub, Cb, Cb);
  for (int i = threadIdx.x; i < 1024; i += blockDim.x) P.s3[(size_t)bp * 1024 + i] = Cb[i];
  for (int idx = threadIdx.x; idx < 16 * 3; idx += blockDim.x) {
    int r = idx / 3, o = idx - r * 3;
    float s = P.outB[o];
    for (int k = 0; k < 64; ++k) s += Cb[r * 64 + k] * P.outW[k * 3 + o];
    P.out[((((size_t)b * T_STEPS + t) * NP + p) * NJ + (6 + r)) * 3 + o] = s;
  }
}

// ---------------- host ----------------
extern "C" void kernel_launch(void* const* d_in, const int* in_sizes, int n_in,
                              void* d_out, int out_size, void* d_ws, size_t ws_size,
                              hipStream_t stream) {
  DevParams P;
  P.xs = (const float*)d_in[0];
  P.ms = (const float*)d_in[1];
  P.h0 = (const float*)d_in[2];
  P.s_adj1 = (const float*)d_in[3];
  P.s_adj2 = (const float*)d_in[4];
  P.s_adj3 = (const float*)d_in[5];
  P.d_adj1 = (const float*)d_in[6];
  P.d_adj2 = (const float*)d_in[7];
  P.d_adj3 = (const float*)d_in[8];
  // params pytree leaves (jax sorted-key order): gru1,gru2,gru3 (each:
  // block0 {Wc,Wr,Wu,bc,br,bu}, block1 same), then ode1..3 {Ws[0..3], bs[0..3]},
  // then out {W,b}
  const int base = 9;
  for (int g = 0; g < 3; ++g)
    for (int blk = 0; blk < 2; ++blk) {
      int o = base + g * 12 + blk * 6;
      P.gruW[g][blk][0] = (const float*)d_in[o + 1]; // Wr
      P.gruW[g][blk][1] = (const float*)d_in[o + 2]; // Wu
      P.gruW[g][blk][2] = (const float*)d_in[o + 0]; // Wc
      P.gruB[g][blk][0] = (const float*)d_in[o + 4]; // br
      P.gruB[g][blk][1] = (const float*)d_in[o + 5]; // bu
      P.gruB[g][blk][2] = (const float*)d_in[o + 3]; // bc
    }
  const int ob = base + 36;
  for (int L = 0; L < 3; ++L)
    for (int l = 0; l < 4; ++l) {
      P.odeW[L][l] = (const float*)d_in[ob + L * 8 + l];
      P.odeB[L][l] = (const float*)d_in[ob + L * 8 + 4 + l];
    }
  P.outW = (const float*)d_in[base + 60];
  P.outB = (const float*)d_in[base + 61];
  P.wa = (_Float16*)d_ws;
  char* fbase = (char*)d_ws + 524288; // past f16 arena (421,888 B)
  P.s1 = (float*)fbase;
  P.s2 = (float*)(fbase + 65536);
  P.s3 = (float*)(fbase + 65536 + 327680);
  P.out = (float*)d_out;

  const size_t SM_ODE1  = 48u * 1024;
  const size_t SM_ODE23 = 112u * 1024;
  const size_t SM_GRU1  = 124u * 1024;
  const size_t SM_GRU2  = 148u * 1024;
  const size_t SM_GRU3  = 150u * 1024;

  k_wconv<<<30, 256, 0, stream>>>(P);
  k_init<<<176, 256, 0, stream>>>(P);
  for (int t = 0; t < T_STEPS; ++t) {
    k_ode1<<<NB, 128, SM_ODE1, stream>>>(P);
    k_ode23<<<NB * NP, 128, SM_ODE23, stream>>>(P);
    k_gru1<<<NB, 128, SM_GRU1, stream>>>(P, t);
    k_gru2<<<NB * NP, 128, SM_GRU2, stream>>>(P, t);
    k_gru3<<<NB * NP, 128, SM_GRU3, stream>>>(P, t);
  }
}
